// iSpikformer_317827580140
// MI455X (gfx1250) — matrix operand order repaired and checked by reading, not compiled, not run on hardware
//
#include <hip/hip_runtime.h>

// ---------------- model constants ----------------
#define BB   16
#define LL   96
#define CC   321
#define DD   512
#define DFFN 2048
#define TT   4
#define HH   8
#define DHD  64
#define BC   (BB * CC)            // 5136  (multiple of 16)
#define BCD  ((size_t)BC * DD)    // 2,629,632

typedef __attribute__((ext_vector_type(16))) _Float16 v16h;
typedef __attribute__((ext_vector_type(8)))  _Float16 v8h;
typedef __attribute__((ext_vector_type(8)))  float    v8f;

__device__ inline v16h mk16(v8h lo, v8h hi) {
  v16h r;
#pragma unroll
  for (int j = 0; j < 8; ++j) { r[j] = lo[j]; r[j + 8] = hi[j]; }
  return r;
}

// -------- CDNA5 async global->LDS (ASYNCcnt-tracked DMA, no VGPR bounce) ----
__device__ inline unsigned lds_off(const void* p) {
  // generic pointers to LDS carry the byte offset in addr[31:0] (aperture rule)
  return (unsigned)(unsigned long long)p;
}
__device__ inline void async_ld_b128(unsigned dst_lds, const void* gsrc) {
  asm volatile("global_load_async_to_lds_b128 %0, %1, off"
               :: "v"(dst_lds), "v"((unsigned long long)gsrc)
               : "memory");
}
__device__ inline void wait_async0() {
  asm volatile("s_wait_asynccnt 0x0" ::: "memory");
}

// ---------------- fp32 [N][K] -> f16 [K][N] weight convert+transpose --------
__global__ void cvt_f16_t(const float* __restrict__ src, _Float16* __restrict__ dst,
                          int N, int K) {
  int i = blockIdx.x * blockDim.x + threadIdx.x;  // output index over [K][N]
  if (i >= N * K) return;
  int k = i / N, n = i - k * N;
  dst[i] = (_Float16)src[(size_t)n * K + k];
}

// ---------------- conv(1,3) + BN + LIF over T ----------------
__global__ void conv_lif(const float* __restrict__ x, const float* __restrict__ cw,
                         const float* __restrict__ cb, const float* __restrict__ cs,
                         const float* __restrict__ csb, _Float16* __restrict__ spk) {
  int i = blockIdx.x * blockDim.x + threadIdx.x;
  if (i >= BB * CC * LL) return;
  int l = i % LL;
  int c = (i / LL) % CC;
  int b = i / (LL * CC);
  float x0 = (l > 0)      ? x[((size_t)b * LL + (l - 1)) * CC + c] : 0.f;
  float x1 =                x[((size_t)b * LL + l) * CC + c];
  float x2 = (l < LL - 1) ? x[((size_t)b * LL + (l + 1)) * CC + c] : 0.f;
  float vm = 0.f;
#pragma unroll
  for (int t = 0; t < TT; ++t) {
    float e = x0 * cw[t * 3 + 0] + x1 * cw[t * 3 + 1] + x2 * cw[t * 3 + 2] + cb[t];
    e = e * cs[t] + csb[t];
    vm = 0.5f * (vm + e);
    float sp = (vm >= 1.0f) ? 1.f : 0.f;
    vm *= (1.f - sp);
    spk[(((size_t)t * BB + b) * CC + c) * LL + l] = (_Float16)sp;
  }
}

// ---------------- fused GEMM + BN + multi-step LIF (+residual) ----------------
// A: [T*R][K] f16 spikes.  W: [K][N] f16 (pre-transposed).  out: [T*R][N].
// grid = (R/16, N/128), block = 128.  Wave w owns time-step t=w; the LIF
// T-scan runs from LDS after the K loop.  Tiles are staged with
// GLOBAL_LOAD_ASYNC_TO_LDS_B128 (requires R % 16 == 0, K % 32 == 0,
// N % 128 == 0 -- true for every launch below, so no zero-fill needed).
// Inner loop: 4 B-fragments loaded into distinct registers, then 4 WMMAs
// issued back-to-back (x2 groups) to keep the matrix pipe busy.
__global__ __launch_bounds__(128) void gemm_bn_lif(
    const _Float16* __restrict__ A, const _Float16* __restrict__ W,
    const float* __restrict__ bias, const float* __restrict__ bns,
    const float* __restrict__ bnb, const _Float16* res, _Float16* out,
    int R, int K, int N, float vth) {
  __shared__ __align__(16) _Float16 As[TT][16][40];
  __shared__ __align__(16) _Float16 Bs[32][136];
  __shared__ __align__(16) float    Ps[TT][16][128];

  const int tid  = threadIdx.x;
  const int wv   = tid >> 5;          // == time step t
  const int lane = tid & 31;
  const int r0   = blockIdx.x * 16;
  const int n0   = blockIdx.y * 128;

  const int larow = lane & 15;
  const int lkh   = (lane >> 4) * 8;            // A K-half select
  const int lkr   = larow + ((lane >> 4) << 4); // B K-row (lane = K row)

  v8f zero = {};
  v8f acc[8];
#pragma unroll
  for (int i = 0; i < 8; ++i) acc[i] = zero;

  for (int k0 = 0; k0 < K; k0 += 32) {
    { // A tile: wave wv DMAs its own 16x32 (t = wv) straight into LDS
      int row = lane >> 1;
      int ch  = (lane & 1) << 4;
      const _Float16* src = A + ((size_t)(wv * R + r0 + row)) * K + k0 + ch;
      async_ld_b128(lds_off(&As[wv][row][ch]), src);
      async_ld_b128(lds_off(&As[wv][row][ch + 8]), src + 8);
    }
    { // B tile: weights are K-major, rows copy straight in (no transpose)
      int row = tid >> 2;
      int seg = (tid & 3) << 5;
      const _Float16* src = W + (size_t)(k0 + row) * N + n0 + seg;
#pragma unroll
      for (int j = 0; j < 4; ++j)
        async_ld_b128(lds_off(&Bs[row][seg + 8 * j]), src + 8 * j);
      if (k0 + 32 < K) __builtin_prefetch(src + (size_t)32 * N, 0, 1);
    }
    wait_async0();
    __syncthreads();

    v8h alo = *(const v8h*)&As[wv][larow][lkh];
    v8h ahi = *(const v8h*)&As[wv][larow][16 + lkh];
    v16h af = mk16(alo, ahi);
#pragma unroll
    for (int g = 0; g < 2; ++g) {
      v16h bfr[4];
#pragma unroll
      for (int nf = 0; nf < 4; ++nf) {
        int col = (g * 4 + nf) * 16;
#pragma unroll
        for (int e = 0; e < 16; ++e)
          bfr[nf][e] = Bs[(e < 8) ? (lkh + e) : (8 + lkh + e)][col + larow];
      }
#pragma unroll
      for (int nf = 0; nf < 4; ++nf)
        acc[g * 4 + nf] = __builtin_amdgcn_wmma_f32_16x16x32_f16(
            false, af, false, bfr[nf], (short)0, acc[g * 4 + nf], false, false);
    }
    __syncthreads();
  }

  // stash preacts for all T in LDS
#pragma unroll
  for (int nf = 0; nf < 8; ++nf) {
#pragma unroll
    for (int vr = 0; vr < 8; ++vr) {
      int rrow = vr + ((lane >> 4) << 3);
      Ps[wv][rrow][nf * 16 + larow] = acc[nf][vr];
    }
  }
  __syncthreads();

  // fused BN affine + LIF scan over T (+ optional residual), write f16 spikes
#pragma unroll
  for (int e = 0; e < 16; ++e) {
    int idx = e * 128 + tid;       // threads stride cols -> coalesced stores
    int row = idx >> 7;
    int col = idx & 127;
    int rg  = r0 + row;
    if (rg >= R) continue;
    int n = n0 + col;
    float bv = bias[n], sv = bns[n], hv = bnb[n];
    float vm = 0.f;
#pragma unroll
    for (int t = 0; t < TT; ++t) {
      float p = (Ps[t][row][col] + bv) * sv + hv;
      vm = 0.5f * (vm + p);
      float sp = (vm >= vth) ? 1.f : 0.f;
      vm *= (1.f - sp);
      size_t o = ((size_t)(t * R + rg)) * N + n;
      float ov = sp;
      if (res) ov += (float)res[o];
      out[o] = (_Float16)ov;
    }
  }
}

// ---------------- fused softmax-free attention: O = (Q K^T * 0.125) V ------
// grid = (T*B*H, ceil(C/32)), block = 128 (4 waves in a 2x2 tile grid).
__global__ __launch_bounds__(128) void attn_qkv(
    const _Float16* __restrict__ q, const _Float16* __restrict__ k,
    const _Float16* __restrict__ v, float* __restrict__ o) {
  __shared__ __align__(16) _Float16 Qs[32][72];
  __shared__ __align__(16) _Float16 Ks[32][72];
  __shared__ __align__(16) _Float16 Vs[32][72];
  __shared__ __align__(16) _Float16 Ss[32][40];

  const int tid  = threadIdx.x;
  const int wv   = tid >> 5;
  const int lane = tid & 31;
  const int mi   = wv >> 1, ni = wv & 1;
  const int tbh  = blockIdx.x;
  const int hh   = tbh & (HH - 1);
  const int tb   = tbh >> 3;
  const int c0   = blockIdx.y * 32;
  const size_t base = (size_t)tb * CC * DD + (size_t)hh * DHD;

  const int larow = lane & 15;
  const int lkh   = (lane >> 4) * 8;
  const int lkr   = larow + ((lane >> 4) << 4);

  { // Q tile (32 rows x 64), zero-filled past C
    int row = tid >> 2;
    int seg = (tid & 3) << 4;
    int cg  = c0 + row;
    v8h lo = {}, hi = {};
    if (cg < CC) {
      const _Float16* src = q + base + (size_t)cg * DD + seg;
      lo = *(const v8h*)src;
      hi = *(const v8h*)(src + 8);
    }
    *(v8h*)&Qs[row][seg]     = lo;
    *(v8h*)&Qs[row][seg + 8] = hi;
  }

  v8f zero = {};
  v8f oacc[2];
  oacc[0] = zero; oacc[1] = zero;

  for (int j0 = 0; j0 < CC; j0 += 32) {
    __syncthreads();
    { // K/V tiles, zero-filled past C
      int row = tid >> 2;
      int seg = (tid & 3) << 4;
      int cg  = j0 + row;
      v8h kl = {}, kh2 = {}, vl = {}, vh2 = {};
      if (cg < CC) {
        const _Float16* ks = k + base + (size_t)cg * DD + seg;
        const _Float16* vs = v + base + (size_t)cg * DD + seg;
        kl = *(const v8h*)ks; kh2 = *(const v8h*)(ks + 8);
        vl = *(const v8h*)vs; vh2 = *(const v8h*)(vs + 8);
      }
      *(v8h*)&Ks[row][seg] = kl; *(v8h*)&Ks[row][seg + 8] = kh2;
      *(v8h*)&Vs[row][seg] = vl; *(v8h*)&Vs[row][seg + 8] = vh2;
    }
    __syncthreads();

    // S quadrant (mi, ni): 16x16, K-dim = DH = 64
    v8f sacc = zero;
#pragma unroll
    for (int kk = 0; kk < DHD; kk += 32) {
      v8h alo = *(const v8h*)&Qs[16 * mi + larow][kk + lkh];
      v8h ahi = *(const v8h*)&Qs[16 * mi + larow][kk + 16 + lkh];
      v16h af = mk16(alo, ahi);
      v16h bf = mk16(*(const v8h*)&Ks[16 * ni + larow][kk + lkh],
                     *(const v8h*)&Ks[16 * ni + larow][kk + 16 + lkh]);
      sacc = __builtin_amdgcn_wmma_f32_16x16x32_f16(false, af, false, bf,
                                                    (short)0, sacc, false, false);
    }
#pragma unroll
    for (int vr = 0; vr < 8; ++vr)
      Ss[16 * mi + vr + ((lane >> 4) << 3)][16 * ni + larow] =
          (_Float16)(sacc[vr] * 0.125f);
    __syncthreads();

    // O (rows 16*mi, cols 32*ni..+31) += S(16x32) @ V(32x64)
    {
      v8h alo = *(const v8h*)&Ss[16 * mi + larow][lkh];
      v8h ahi = *(const v8h*)&Ss[16 * mi + larow][16 + lkh];
      v16h af = mk16(alo, ahi);
      v16h bfr[2];
#pragma unroll
      for (int nf = 0; nf < 2; ++nf) {
#pragma unroll
        for (int e = 0; e < 16; ++e)
          bfr[nf][e] = Vs[(e < 8) ? (lkh + e) : (8 + lkh + e)][32 * ni + nf * 16 + larow];
      }
#pragma unroll
      for (int nf = 0; nf < 2; ++nf)
        oacc[nf] = __builtin_amdgcn_wmma_f32_16x16x32_f16(
            false, af, false, bfr[nf], (short)0, oacc[nf], false, false);
    }
  }

#pragma unroll
  for (int nf = 0; nf < 2; ++nf) {
#pragma unroll
    for (int vr = 0; vr < 8; ++vr) {
      int rr = 16 * mi + vr + ((lane >> 4) << 3);
      int cg = c0 + rr;
      if (cg < CC) {
        int dd = 32 * ni + nf * 16 + larow;
        o[base + (size_t)cg * DD + dd] = oacc[nf][vr];
      }
    }
  }
}

// ---------------- plain LIF over T (attention output, thr/2) ----------------
__global__ void lif_plain(const float* __restrict__ pre, _Float16* __restrict__ out,
                          int inner, float vth) {
  int i = blockIdx.x * blockDim.x + threadIdx.x;
  if (i >= inner) return;
  float vm = 0.f;
#pragma unroll
  for (int t = 0; t < TT; ++t) {
    float p = pre[(size_t)t * inner + i];
    vm = 0.5f * (vm + p);
    float sp = (vm >= vth) ? 1.f : 0.f;
    vm *= (1.f - sp);
    out[(size_t)t * inner + i] = (_Float16)sp;
  }
}

// ---------------- finalize: out = out2 = h[T-1] as fp32 ----------------
__global__ void final_out(const _Float16* __restrict__ h, float* __restrict__ out, int n) {
  int i = blockIdx.x * blockDim.x + threadIdx.x;
  if (i >= n) return;
  float v = (float)h[(size_t)(TT - 1) * n + i];
  out[i] = v;
  out[(size_t)n + i] = v;
}

// ---------------------------------------------------------------------------
extern "C" void kernel_launch(void* const* d_in, const int* in_sizes, int n_in,
                              void* d_out, int out_size, void* d_ws, size_t ws_size,
                              hipStream_t stream) {
  (void)in_sizes; (void)n_in; (void)out_size; (void)ws_size;

  const float* X       = (const float*)d_in[0];
  const float* conv_w  = (const float*)d_in[1];
  const float* conv_b  = (const float*)d_in[2];
  const float* conv_s  = (const float*)d_in[3];
  const float* conv_sb = (const float*)d_in[4];
  const float* emb_w   = (const float*)d_in[5];
  const float* emb_b   = (const float*)d_in[6];
  const float* emb_s   = (const float*)d_in[7];
  const float* emb_sb  = (const float*)d_in[8];

  // ---- workspace layout (all region sizes are multiples of 256B) ----
  char* ws = (char*)d_ws;
  const size_t W16_BYTES  = 6340608ull * 2;                 // 12,681,216
  const size_t SPKL_BYTES = (size_t)TT * BB * CC * LL * 2;  //  3,944,448
  const size_t H_BYTES    = (size_t)TT * BCD * 2;           // 21,037,056
  _Float16* W16  = (_Float16*)ws;
  _Float16* spkL = (_Float16*)(ws + W16_BYTES);
  _Float16* hbuf = (_Float16*)(ws + W16_BYTES + SPKL_BYTES);
  char* scr      = ws + W16_BYTES + SPKL_BYTES + H_BYTES;
  _Float16* qb   = (_Float16*)scr;                  // also reused as attnS
  _Float16* kb   = (_Float16*)(scr + H_BYTES);
  _Float16* vb   = (_Float16*)(scr + 2 * H_BYTES);
  float*    aO   = (float*)(scr + 3 * H_BYTES);     // [T,BC,D] fp32 preacts
  _Float16* aS   = qb;                              // attention spikes (q dead)
  _Float16* mb   = (_Float16*)scr;                  // [T,BC,DFF] (q/k/v/aO dead)

  auto cvt = [&](const float* src, _Float16* dst, int N, int K) {
    size_t n = (size_t)N * K;
    cvt_f16_t<<<dim3((unsigned)((n + 255) / 256)), dim3(256), 0, stream>>>(src, dst, N, K);
  };
  auto gemm = [&](const _Float16* Ap, const _Float16* Wp, const float* b,
                  const float* s, const float* sb, const _Float16* resp,
                  _Float16* outp, int K, int N, float vth) {
    dim3 grid(BC / 16, N / 128);
    gemm_bn_lif<<<grid, dim3(128), 0, stream>>>(Ap, Wp, b, s, sb, resp, outp,
                                                BC, K, N, vth);
  };

  // ---- weight conversion to f16, transposed to [K][N] for direct staging ----
  const size_t BLK_W = 3145728;  // per-block f16 weight footprint (elems)
  cvt(emb_w, W16, DD, LL);
  for (int bk = 0; bk < 2; ++bk) {
    int base = 9 + bk * 24;
    _Float16* wb = W16 + 49152 + (size_t)bk * BLK_W;
    cvt((const float*)d_in[base + 0],  wb + 0,       DD,   DD);    // q
    cvt((const float*)d_in[base + 4],  wb + 262144,  DD,   DD);    // k
    cvt((const float*)d_in[base + 8],  wb + 524288,  DD,   DD);    // v
    cvt((const float*)d_in[base + 12], wb + 786432,  DD,   DD);    // last
    cvt((const float*)d_in[base + 16], wb + 1048576, DFFN, DD);    // fc1
    cvt((const float*)d_in[base + 20], wb + 2097152, DD,   DFFN);  // fc2
  }

  // ---- conv encoder + LIF ----
  conv_lif<<<dim3((BB * CC * LL + 255) / 256), dim3(256), 0, stream>>>(
      X, conv_w, conv_b, conv_s, conv_sb, spkL);

  // ---- inverted embedding (L -> D) + BN + LIF ----
  gemm(spkL, W16, emb_b, emb_s, emb_sb, nullptr, hbuf, LL, DD, 1.0f);

  // ---- transformer blocks ----
  for (int bk = 0; bk < 2; ++bk) {
    int base = 9 + bk * 24;
    _Float16* wb = W16 + 49152 + (size_t)bk * BLK_W;
    const float* qB  = (const float*)d_in[base + 1];
    const float* qS  = (const float*)d_in[base + 2];
    const float* qSb = (const float*)d_in[base + 3];
    const float* kB  = (const float*)d_in[base + 5];
    const float* kS  = (const float*)d_in[base + 6];
    const float* kSb = (const float*)d_in[base + 7];
    const float* vB  = (const float*)d_in[base + 9];
    const float* vS  = (const float*)d_in[base + 10];
    const float* vSb = (const float*)d_in[base + 11];
    const float* lB  = (const float*)d_in[base + 13];
    const float* lS  = (const float*)d_in[base + 14];
    const float* lSb = (const float*)d_in[base + 15];
    const float* f1B  = (const float*)d_in[base + 17];
    const float* f1S  = (const float*)d_in[base + 18];
    const float* f1Sb = (const float*)d_in[base + 19];
    const float* f2B  = (const float*)d_in[base + 21];
    const float* f2S  = (const float*)d_in[base + 22];
    const float* f2Sb = (const float*)d_in[base + 23];

    gemm(hbuf, wb + 0,      qB, qS, qSb, nullptr, qb, DD, DD, 1.0f);
    gemm(hbuf, wb + 262144, kB, kS, kSb, nullptr, kb, DD, DD, 1.0f);
    gemm(hbuf, wb + 524288, vB, vS, vSb, nullptr, vb, DD, DD, 1.0f);

    attn_qkv<<<dim3(TT * BB * HH, (CC + 31) / 32), dim3(128), 0, stream>>>(qb, kb, vb, aO);
    lif_plain<<<dim3((unsigned)((BCD + 255) / 256)), dim3(256), 0, stream>>>(
        aO, aS, (int)BCD, 0.5f);  // attn_lif: threshold thr/2

    gemm(aS,   wb + 786432,  lB,  lS,  lSb,  hbuf,    hbuf, DD,   DD,   1.0f);  // h += o
    gemm(hbuf, wb + 1048576, f1B, f1S, f1Sb, nullptr, mb,   DD,   DFFN, 1.0f);
    gemm(mb,   wb + 2097152, f2B, f2S, f2Sb, hbuf,    hbuf, DFFN, DD,   1.0f);  // h += m
  }

  final_out<<<dim3((unsigned)((BCD + 255) / 256)), dim3(256), 0, stream>>>(
      hbuf, (float*)d_out, (int)BCD);
}
